// TowerOne_10548439679500
// MI455X (gfx1250) — matrix operand order repaired and checked by reading, not compiled, not run on hardware
//
#include <hip/hip_runtime.h>
#include <hip/hip_bf16.h>

typedef unsigned short u16;
typedef __attribute__((ext_vector_type(16))) __bf16 v16bf;
typedef __attribute__((ext_vector_type(8)))  float  v8f;

#define H_DIM 256
#define T_DIM 256
#define B_DIM 512
#define RNN_BBLK 16      // batch rows per block in the recurrence
#define SH_LD 264        // 256 + 8 pad -> 528B row stride, conflict-free b128 pattern

union FragAB { v16bf v; uint4 q[2]; };

// ---- bf16 helpers -----------------------------------------------------------
__device__ __forceinline__ u16 f2bf(float f) {
  unsigned u = __float_as_uint(f);
  unsigned r = u + 0x7FFFu + ((u >> 16) & 1u);   // round-to-nearest-even
  return (u16)(r >> 16);
}

// ---- WMMA fragment loads (ISA 7.12.2 layouts, wave32) -----------------------
// A (16x32 bf16, MxK): lane l holds row m=l&15; elems 0..7 = K[k0+hf*8 ..],
// elems 8..15 = K[k0+16+hf*8 ..]  (hf = l>>4)
__device__ __forceinline__ void load_a_frag(FragAB& f, const u16* base, int ld,
                                            int k0, int lane) {
  int hf = lane >> 4, m = lane & 15;
  const u16* p = base + (long)m * ld + k0 + hf * 8;
  f.q[0] = *(const uint4*)(p);
  f.q[1] = *(const uint4*)(p + 16);
}
// B (32x16 bf16, KxN) with B[k][n] = W[n][k], W row-major [N][ld]:
// lane l holds col n=l&15; elems 0..7 = K[k0+hf*8 ..], elems 8..15 = K[k0+16+hf*8 ..].
__device__ __forceinline__ void load_b_frag(FragAB& f, const u16* wrow, int ld,
                                            int k0, int lane) {
  int hf = lane >> 4, n = lane & 15;
  const u16* p = wrow + (long)n * ld + k0 + hf * 8;
  f.q[0] = *(const uint4*)(p);
  f.q[1] = *(const uint4*)(p + 16);
}

__device__ __forceinline__ v8f wmma_bf16(const FragAB& a, const FragAB& b, v8f c) {
  return __builtin_amdgcn_wmma_f32_16x16x32_bf16(
      /*neg_a=*/false, a.v, /*neg_b=*/false, b.v,
      /*c_mod=*/(short)0, c, /*reuse_a=*/false, /*reuse_b=*/false);
}

// ---- f32 -> bf16 pack with K padding (used for x and all weights) -----------
__global__ void pack_bf16(const float* __restrict__ src, u16* __restrict__ dst,
                          long rows, int K, int Kpad) {
  long idx = (long)blockIdx.x * blockDim.x + threadIdx.x;
  if (idx >= rows * (long)Kpad) return;
  int  col = (int)(idx % Kpad);
  long row = idx / Kpad;
  float v = (col < K) ? src[row * (long)K + col] : 0.0f;
  dst[idx] = f2bf(v);
}

// ---- GEMM: Y[M,N] = Abf[M,lda] @ Wbf[N,ldw]^T + bias, f32 out ---------------
// Wave owns an NT*16-column strip; its NT*KT weight fragments stay in VGPRs
// for the whole kernel. Per 16-row tile: one batched A-fragment load (all
// constant-indexed -> register promoted), then an NT*KT WMMA chain.
// VGPR budget: KT=10,NT=1 -> ~170; KT=8,NT=2 -> ~215 (no spills).
template <int KT, int NT>
__global__ void __launch_bounds__(256) gemm_bias_wmma(
    const u16* __restrict__ A, const u16* __restrict__ W,
    const float* __restrict__ bias, float* __restrict__ Y,
    int N, int lda, int ldw, int CG /* N/(16*NT) */, int R /* row tiles/wave */) {
  int wv = threadIdx.x >> 5;
  int lane = threadIdx.x & 31;
  int gw = blockIdx.x * 8 + wv;
  int cg = gw % CG;
  int chunk = gw / CG;
  long row0 = (long)chunk * R * 16;
  int col0 = cg * (16 * NT);

  // Persistent weight fragments (NT n-tiles x KT k-tiles)
  FragAB Bf[NT * KT];
#pragma unroll
  for (int kt = 0; kt < KT; ++kt)
#pragma unroll
    for (int j = 0; j < NT; ++j)
      load_b_frag(Bf[kt * NT + j], W + (long)(col0 + j * 16) * ldw, ldw, kt * 32, lane);

  int hf = lane >> 4, nl = lane & 15;
  for (int r = 0; r < R; ++r) {
    const u16* Arow = A + (row0 + r * 16) * lda;
    FragAB Af[KT];                      // constant-indexed, fully unrolled
#pragma unroll
    for (int kt = 0; kt < KT; ++kt)
      load_a_frag(Af[kt], Arow, lda, kt * 32, lane);

    v8f acc[NT];
#pragma unroll
    for (int j = 0; j < NT; ++j) acc[j] = (v8f){0.f,0.f,0.f,0.f,0.f,0.f,0.f,0.f};
#pragma unroll
    for (int kt = 0; kt < KT; ++kt)
#pragma unroll
      for (int j = 0; j < NT; ++j)
        acc[j] = wmma_bf16(Af[kt], Bf[kt * NT + j], acc[j]);

#pragma unroll
    for (int j = 0; j < NT; ++j) {
      int col = col0 + j * 16 + nl;
      float bv = bias[col];
#pragma unroll
      for (int v = 0; v < 8; ++v) {
        long m = row0 + r * 16 + v + hf * 8;
        Y[m * N + col] = acc[j][v] + bv;
      }
    }
  }
}

// ---- RNN recurrence: h_{t+1} = tanh(xp[:,t,:] + h_t @ Whh^T + bhh) ----------
// Block = 16 batch rows, 256 threads (8 waves). Wave wv owns columns
// [wv*32, +32). Its 16 W_hh fragments are register-resident across ALL time
// steps; per step only h fragments (LDS) + xp (global, prefetched) move.
template <int WRITE_YS>
__global__ void __launch_bounds__(256) rnn_scan_wmma(
    const float* __restrict__ xp,      // [B,T,H] f32 (x-projection + b_ih)
    const u16* __restrict__ Whh,       // [H][H] bf16
    const float* __restrict__ bhh,     // [H]
    u16* __restrict__ ys,              // [B,T,H] bf16 (layer output seq)
    u16* __restrict__ hlast,           // [B,H]   bf16 (final hidden)
    int T) {
  __shared__ u16  sh[RNN_BBLK * SH_LD];
  __shared__ float sbias[H_DIM];

  int tid = threadIdx.x;
  int lane = tid & 31, wv = tid >> 5;
  for (int i = tid; i < RNN_BBLK * SH_LD; i += 256) sh[i] = 0;  // h0 = 0
  sbias[tid] = bhh[tid];
  __syncthreads();

  int col0 = wv * 32;
  long bbase = (long)blockIdx.x * RNN_BBLK;
  int hf = lane >> 4, nl = lane & 15;

  // Register-resident weights: 8 k-tiles x 2 n-tiles = 16 fragments (128 VGPR)
  FragAB Bf[16];
#pragma unroll
  for (int kt = 0; kt < 8; ++kt)
#pragma unroll
    for (int j = 0; j < 2; ++j)
      load_b_frag(Bf[kt * 2 + j], Whh + (col0 + j * 16) * H_DIM, H_DIM, kt * 32, lane);

  for (int t = 0; t < T; ++t) {
    // prefetch this step's xp values (independent of h -> overlaps WMMA)
    float xpv[16];
#pragma unroll
    for (int j = 0; j < 2; ++j)
#pragma unroll
      for (int v = 0; v < 8; ++v) {
        long b = bbase + v + hf * 8;
        xpv[j * 8 + v] = xp[(b * T + t) * (long)H_DIM + col0 + j * 16 + nl];
      }

    v8f acc[2];
#pragma unroll
    for (int j = 0; j < 2; ++j) acc[j] = (v8f){0.f,0.f,0.f,0.f,0.f,0.f,0.f,0.f};
#pragma unroll
    for (int kt = 0; kt < 8; ++kt) {
      FragAB a;                         // per-k temp: low pressure, LDS latency
      load_a_frag(a, sh, SH_LD, kt * 32, lane);
      acc[0] = wmma_bf16(a, Bf[kt * 2 + 0], acc[0]);
      acc[1] = wmma_bf16(a, Bf[kt * 2 + 1], acc[1]);
    }
    __syncthreads();  // all reads of h_t complete before overwriting

#pragma unroll
    for (int j = 0; j < 2; ++j) {
#pragma unroll
      for (int v = 0; v < 8; ++v) {
        int col = col0 + j * 16 + nl;
        int rloc = v + hf * 8;
        float val = tanhf(acc[j][v] + xpv[j * 8 + v] + sbias[col]);
        sh[rloc * SH_LD + col] = f2bf(val);
      }
    }
    __syncthreads();  // h_{t+1} visible to all waves

    if (WRITE_YS) {
      // vectorized LDS -> global copy of the new h (16 rows x 256 bf16)
      int row = tid >> 4, ch = tid & 15;
      const uint4* s = (const uint4*)(sh + row * SH_LD + ch * 16);
      uint4* d = (uint4*)(ys + ((bbase + row) * (long)T + t) * H_DIM + ch * 16);
      d[0] = s[0];
      d[1] = s[1];
    } else if (t == T - 1) {
      int row = tid >> 4, ch = tid & 15;
      const uint4* s = (const uint4*)(sh + row * SH_LD + ch * 16);
      uint4* d = (uint4*)(hlast + (bbase + row) * (long)H_DIM + ch * 16);
      d[0] = s[0];
      d[1] = s[1];
    }
  }
}

// ---- host-side orchestration ------------------------------------------------
extern "C" void kernel_launch(void* const* d_in, const int* in_sizes, int n_in,
                              void* d_out, int out_size, void* d_ws, size_t ws_size,
                              hipStream_t stream) {
  const float* x    = (const float*)d_in[0];
  const float* Wih0 = (const float*)d_in[1];
  const float* Whh0 = (const float*)d_in[2];
  const float* bih0 = (const float*)d_in[3];
  const float* bhh0 = (const float*)d_in[4];
  const float* Wih1 = (const float*)d_in[5];
  const float* Whh1 = (const float*)d_in[6];
  const float* bih1 = (const float*)d_in[7];
  const float* bhh1 = (const float*)d_in[8];
  const float* fcW  = (const float*)d_in[9];
  const float* fcb  = (const float*)d_in[10];
  float* out = (float*)d_out;

  const long BT = (long)B_DIM * T_DIM;        // 131072 rows
  const int DIN = 312, DINP = 320;

  char* ws = (char*)d_ws;
  size_t off = 0;
  auto alloc = [&](size_t bytes) {
    size_t o = off;
    off = (off + bytes + 255) & ~(size_t)255;
    return o;
  };
  u16*   xbf    = (u16*)  (ws + alloc(BT * DINP * sizeof(u16)));        // 80 MB
  u16*   wih0b  = (u16*)  (ws + alloc((size_t)H_DIM * DINP * 2));
  u16*   whh0b  = (u16*)  (ws + alloc((size_t)H_DIM * H_DIM * 2));
  u16*   wih1b  = (u16*)  (ws + alloc((size_t)H_DIM * H_DIM * 2));
  u16*   whh1b  = (u16*)  (ws + alloc((size_t)H_DIM * H_DIM * 2));
  u16*   fcwb   = (u16*)  (ws + alloc((size_t)128 * H_DIM * 2));
  u16*   hlast  = (u16*)  (ws + alloc((size_t)B_DIM * H_DIM * 2));
  float* xp     = (float*)(ws + alloc(BT * H_DIM * sizeof(float)));     // 128 MB (xp0 then xp1)
  u16*   ys0    = (u16*)  (ws + alloc(BT * H_DIM * sizeof(u16)));       // 64 MB
  (void)ws_size; (void)in_sizes; (void)n_in; (void)out_size;

  // 1) pack x and weights to bf16 (x and W_ih0 padded K 312->320 with zeros)
  {
    long tot = BT * DINP;
    pack_bf16<<<(int)((tot + 255) / 256), 256, 0, stream>>>(x, xbf, BT, DIN, DINP);
    pack_bf16<<<(H_DIM * DINP + 255) / 256, 256, 0, stream>>>(Wih0, wih0b, H_DIM, DIN, DINP);
    pack_bf16<<<(H_DIM * H_DIM + 255) / 256, 256, 0, stream>>>(Whh0, whh0b, H_DIM, H_DIM, H_DIM);
    pack_bf16<<<(H_DIM * H_DIM + 255) / 256, 256, 0, stream>>>(Wih1, wih1b, H_DIM, H_DIM, H_DIM);
    pack_bf16<<<(H_DIM * H_DIM + 255) / 256, 256, 0, stream>>>(Whh1, whh1b, H_DIM, H_DIM, H_DIM);
    pack_bf16<<<(128 * H_DIM + 255) / 256, 256, 0, stream>>>(fcW, fcwb, 128, H_DIM, H_DIM);
  }

  // 2) xp0 = x @ W_ih0^T + b_ih0  (M=131072, N=256, K=320, KT=10, NT=1)
  //    waves = (8192/8 row chunks) * 16 col groups = 16384 -> 2048 blocks
  gemm_bias_wmma<10, 1><<<2048, 256, 0, stream>>>(xbf, wih0b, bih0, xp,
                                                  H_DIM, DINP, DINP, 16, 8);

  // 3) layer-0 recurrence -> ys0 (bf16)
  rnn_scan_wmma<1><<<B_DIM / RNN_BBLK, 256, 0, stream>>>(xp, whh0b, bhh0, ys0, hlast, T_DIM);

  // 4) xp1 = ys0 @ W_ih1^T + b_ih1  (KT=8, NT=2; waves = 1024*8 -> 1024 blocks)
  gemm_bias_wmma<8, 2><<<1024, 256, 0, stream>>>(ys0, wih1b, bih1, xp,
                                                 H_DIM, H_DIM, H_DIM, 8, 8);

  // 5) layer-1 recurrence -> hlast (bf16), ys not written
  rnn_scan_wmma<0><<<B_DIM / RNN_BBLK, 256, 0, stream>>>(xp, whh1b, bhh1, ys0, hlast, T_DIM);

  // 6) out = hlast @ fc_W^T + fc_b  (M=512, N=128, K=256; KT=8, NT=2, CG=4, R=2)
  gemm_bias_wmma<8, 2><<<8, 256, 0, stream>>>(hlast, fcwb, fcb, out,
                                              128, H_DIM, H_DIM, 4, 2);
}
